// DeepFM_35588099014965
// MI455X (gfx1250) — matrix operand order repaired and checked by reading, not compiled, not run on hardware
//
#include <hip/hip_runtime.h>
#include <stdint.h>

typedef unsigned short u16;
typedef __attribute__((ext_vector_type(16))) __bf16 v16bf;
typedef __attribute__((ext_vector_type(8)))  float  v8f;

#define B_ROWS 16384
#define F_DIM  512
#define MT     64     // rows per block
#define TPB    256    // 8 waves: wave pairs split N-tiles, same LDS tile
#define PAD    8

// LDS row strides (elements)
#define XS   (F_DIM + PAD)   // 520
#define W1S  (F_DIM + PAD)
#define FMS  (F_DIM + PAD)
#define H1S  (128 + PAD)     // 136
#define W2S  (128 + PAD)
#define H2S  (64 + PAD)      // 72
#define W3S  (64 + PAD)
#define H3S  (32 + PAD)      // 40

// d_ws layout (bf16/u16 element offsets), all [N][K] row-major transposed weights
#define WS_W1T   0                       // 128 x 512
#define WS_W2T   (WS_W1T + 128 * 512)    // 64 x 128
#define WS_W3T   (WS_W2T + 64 * 128)     // 32 x 64
#define WS_FMB   (WS_W3T + 32 * 64)      // 32 x 512  rows 0-15: fm_w^T ; 16-31: (2*fm_w*fm_b)^T
#define WS_FMW2  (WS_FMB + 32 * 512)     // 16 x 512  (fm_w^2)^T
#define WS_BF_TOTAL (WS_FMW2 + 16 * 512) // 100352 u16
// float region right after: SFB[16] = sum_f fm_b ; SFB2[16] = sum_f fm_b^2

#define SMEM_USHORTS (MT*XS + 128*W1S + 32*FMS + 16*FMS + MT*H1S + 64*W2S + MT*H2S + 32*W3S + MT*H3S)
#define SMEM_FLOATS  (MT*32 + MT*16 + TPB)
#define SMEM_BYTES   (SMEM_USHORTS*2 + SMEM_FLOATS*4)   // 316672 B <= 320KB

static __device__ __forceinline__ u16 f2bf(float f) {
  unsigned u = __float_as_uint(f);
  unsigned r = (u + 0x7FFFu + ((u >> 16) & 1u)) >> 16;
  return (u16)r;
}
static __device__ __forceinline__ float bf2f(u16 h) {
  return __uint_as_float(((unsigned)h) << 16);
}

struct Frag { union { v16bf v; uint4 q[2]; u16 s[16]; }; };

// A fragment: 16x32 bf16, M = lane%16, elems 0..7 -> K=k0+half*8+e, elems 8..15 -> K=k0+16+half*8+e
static __device__ __forceinline__ v16bf load_a(const u16* base, int stride, int m0, int k0, int lane) {
  const int half = lane >> 4, r = lane & 15;
  const u16* p = base + (m0 + r) * stride + k0 + half * 8;
  Frag f;
  f.q[0] = *(const uint4*)(p);
  f.q[1] = *(const uint4*)(p + 16);
  return f.v;
}
static __device__ __forceinline__ v16bf square_frag(v16bf in) {
  Frag f; f.v = in;
#pragma unroll
  for (int i = 0; i < 16; ++i) { float t = bf2f(f.s[i]); f.s[i] = f2bf(t * t); }
  return f.v;
}
// B fragment from [N][K]-major LDS: N = lane%16, elems 0..7 -> K=k0+half*8+e, elems 8..15 -> K=k0+16+half*8+e
static __device__ __forceinline__ v16bf load_b(const u16* base, int stride, int n0, int k0, int lane) {
  const int half = lane >> 4, r = lane & 15;
  const u16* p = base + (n0 + r) * stride + k0 + half * 8;
  Frag f;
  f.q[0] = *(const uint4*)(p);
  f.q[1] = *(const uint4*)(p + 16);
  return f.v;
}

// Software-pipelined GEMM: prefetch next k-step's fragments before issuing WMMA k,
// so the backend can emit partial s_wait_dscnt and overlap DS loads with matrix math.
template<int NT, int KS>
static __device__ __forceinline__ void gemm_relu_bf16(const u16* A, int As, const u16* Bm, int Bs,
                                                      const float* bias, u16* Out, int Os,
                                                      int m0, int nt_base, int lane) {
  const int half = lane >> 4, r = lane & 15;
#pragma unroll
  for (int nt = 0; nt < NT; ++nt) {
    const int n0 = (nt_base + nt) * 16;
    v8f acc;
#pragma unroll
    for (int i = 0; i < 8; ++i) acc[i] = 0.0f;
    v16bf a = load_a(A, As, m0, 0, lane);
    v16bf b = load_b(Bm, Bs, n0, 0, lane);
#pragma unroll
    for (int ks = 0; ks < KS; ++ks) {
      v16bf an = a, bn = b;
      if (ks + 1 < KS) {
        an = load_a(A, As, m0, (ks + 1) * 32, lane);
        bn = load_b(Bm, Bs, n0, (ks + 1) * 32, lane);
      }
      acc = __builtin_amdgcn_wmma_f32_16x16x32_bf16(false, a, false, b, (short)0, acc, false, false);
      a = an; b = bn;
    }
    float bv = bias[n0 + r];
#pragma unroll
    for (int v = 0; v < 8; ++v) {
      float o = fmaxf(acc[v] + bv, 0.0f);
      Out[(m0 + v + 8 * half) * Os + n0 + r] = f2bf(o);
    }
  }
}

template<int NT, int KS, bool SQ>
static __device__ __forceinline__ void gemm_f32_out(const u16* A, int As, const u16* Bm, int Bs,
                                                    float* Out, int Os, int m0, int nt_base, int lane) {
  const int half = lane >> 4, r = lane & 15;
#pragma unroll
  for (int nt = 0; nt < NT; ++nt) {
    const int n0 = (nt_base + nt) * 16;
    v8f acc;
#pragma unroll
    for (int i = 0; i < 8; ++i) acc[i] = 0.0f;
    v16bf a = load_a(A, As, m0, 0, lane);
    v16bf b = load_b(Bm, Bs, n0, 0, lane);
#pragma unroll
    for (int ks = 0; ks < KS; ++ks) {
      v16bf an = a, bn = b;
      if (ks + 1 < KS) {
        an = load_a(A, As, m0, (ks + 1) * 32, lane);
        bn = load_b(Bm, Bs, n0, (ks + 1) * 32, lane);
      }
      v16bf au = SQ ? square_frag(a) : a;
      acc = __builtin_amdgcn_wmma_f32_16x16x32_bf16(false, au, false, b, (short)0, acc, false, false);
      a = an; b = bn;
    }
#pragma unroll
    for (int v = 0; v < 8; ++v)
      Out[(m0 + v + 8 * half) * Os + n0 + r] = acc[v];
  }
}

// Issue async global->LDS b128 copies (ASYNCcnt path). g is [rows][kelems] dense, l is [rows][lstride].
static __device__ __forceinline__ void async_stage(const u16* g, u16* l, int rows, int kelems,
                                                   int lstride, int tid, const char* lds_base) {
  const int cpr = kelems >> 3;            // 16B chunks per row
  const int chunks = rows * cpr;
  for (int c = tid; c < chunks; c += TPB) {
    int r = c / cpr, kc = (c - r * cpr) << 3;
    unsigned lds_off = (unsigned)((const char*)(l + r * lstride + kc) - lds_base);
    unsigned long long ga = (unsigned long long)(uintptr_t)(g + r * kelems + kc);
    asm volatile("global_load_async_to_lds_b128 %0, %1, off"
                 :: "v"(lds_off), "v"(ga) : "memory");
  }
}

// ---------------- prep: transpose + convert weights to bf16 in d_ws ----------------
__global__ void deepfm_prep(const float* __restrict__ fm_w, const float* __restrict__ fm_b,
                            const float* __restrict__ w1, const float* __restrict__ w2,
                            const float* __restrict__ w3,
                            u16* __restrict__ wsb, float* __restrict__ wsf) {
  int i = blockIdx.x * blockDim.x + threadIdx.x;
  if (i < WS_W2T) {                               // W1T[n][k] = w1[k][n]
    int n = i / 512, k = i - n * 512;
    wsb[i] = f2bf(w1[k * 128 + n]);
  } else if (i < WS_W3T) {                        // W2T[n][k] = w2[k][n]
    int j = i - WS_W2T; int n = j / 128, k = j - n * 128;
    wsb[i] = f2bf(w2[k * 64 + n]);
  } else if (i < WS_FMB) {                        // W3T[n][k] = w3[k][n]
    int j = i - WS_W3T; int n = j / 64, k = j - n * 64;
    wsb[i] = f2bf(w3[k * 32 + n]);
  } else if (i < WS_FMW2) {                       // FMB rows 0-15: fm_w^T ; 16-31: 2*fm_w*fm_b
    int j = i - WS_FMB; int rr = j / 512, f = j - rr * 512;
    float v = (rr < 16) ? fm_w[f * 16 + rr]
                        : 2.0f * fm_w[f * 16 + (rr - 16)] * fm_b[f * 16 + (rr - 16)];
    wsb[i] = f2bf(v);
  } else if (i < WS_BF_TOTAL) {                   // FMW2[d][f] = fm_w[f][d]^2
    int j = i - WS_FMW2; int d = j / 512, f = j - d * 512;
    float v = fm_w[f * 16 + d];
    wsb[i] = f2bf(v * v);
  }
  if (i < 32) {                                   // SFB / SFB2
    int d = i & 15;
    float s = 0.0f;
    for (int f = 0; f < 512; ++f) {
      float v = fm_b[f * 16 + d];
      s += (i < 16) ? v : v * v;
    }
    wsf[i] = s;
  }
}

// ---------------- fused DeepFM forward ----------------
__global__ __launch_bounds__(TPB) void deepfm_main(
    const float* __restrict__ x, const float* __restrict__ lin_w, const float* __restrict__ lin_b,
    const float* __restrict__ b1, const float* __restrict__ b2, const float* __restrict__ b3,
    const float* __restrict__ wo, const float* __restrict__ bo, const float* __restrict__ bias0,
    const u16* __restrict__ wsb, const float* __restrict__ wsf, float* __restrict__ out) {
  extern __shared__ char smem[];
  u16*  XB    = (u16*)smem;                 // 64 x 520
  u16*  W1T   = XB   + MT * XS;             // 128 x 520
  u16*  FMB   = W1T  + 128 * W1S;           // 32 x 520
  u16*  FMW2  = FMB  + 32 * FMS;            // 16 x 520
  u16*  H1    = FMW2 + 16 * FMS;            // 64 x 136
  u16*  W2T   = H1   + MT * H1S;            // 64 x 136
  u16*  H2    = W2T  + 64 * W2S;            // 64 x 72
  u16*  W3T   = H2   + MT * H2S;            // 32 x 72
  u16*  H3    = W3T  + 32 * W3S;            // 64 x 40
  float* FMOUT = (float*)(H3 + MT * H3S);   // 64 x 32
  float* QOUT  = FMOUT + MT * 32;           // 64 x 16
  float* LINP  = QOUT + MT * 16;            // 256

  const int tid  = threadIdx.x;
  const int lane = tid & 31;
  const int w    = tid >> 5;                // wave 0..7
  const int m0   = (w & 3) * 16;            // wave's 16-row stripe
  const int nh   = w >> 2;                  // which N-half this wave covers
  const int row0 = blockIdx.x * MT;

  // 1) issue all weight stages as async global->LDS DMA (overlaps with x staging below)
  async_stage(wsb + WS_W1T,  W1T,  128, 512, W1S, tid, smem);
  async_stage(wsb + WS_FMB,  FMB,  32,  512, FMS, tid, smem);
  async_stage(wsb + WS_FMW2, FMW2, 16,  512, FMS, tid, smem);
  async_stage(wsb + WS_W2T,  W2T,  64,  128, W2S, tid, smem);
  async_stage(wsb + WS_W3T,  W3T,  32,  64,  W3S, tid, smem);

  // 2) x tile: f32 -> bf16 into LDS; fold the linear term (f32 accuracy) while DMA is in flight
  {
    int s = tid & 3, row = tid >> 2;        // 4 threads per row, 128 cols each
    const float* xr = x + (size_t)(row0 + row) * F_DIM + s * 128;
    const float* lw = lin_w + s * 128;
    u16* dst = XB + row * XS + s * 128;
    float acc = 0.0f;
#pragma unroll 4
    for (int c = 0; c < 128; c += 4) {
      float4 xv = *(const float4*)(xr + c);
      float4 wv = *(const float4*)(lw + c);
      acc = fmaf(xv.x, wv.x, acc); acc = fmaf(xv.y, wv.y, acc);
      acc = fmaf(xv.z, wv.z, acc); acc = fmaf(xv.w, wv.w, acc);
      dst[c + 0] = f2bf(xv.x); dst[c + 1] = f2bf(xv.y);
      dst[c + 2] = f2bf(xv.z); dst[c + 3] = f2bf(xv.w);
    }
    LINP[tid] = acc;
  }
  asm volatile("s_wait_asynccnt 0x0" ::: "memory");
  __syncthreads();

  // 3) GEMMs (branching is wave-uniform -> EXEC all ones for every WMMA)
  //    deep L1: H1 = relu(XB @ w1 + b1)   [64,512]x[512,128]; wave covers 4 of 8 N-tiles
  gemm_relu_bf16<4, 16>(XB, XS, W1T, W1S, b1, H1, H1S, m0, nh * 4, lane);
  //    FM: half the waves do [x@fm_w | x@(2 fm_w fm_b)], the others do (x^2)@(fm_w^2)
  if (nh == 0) {
    gemm_f32_out<2, 16, false>(XB, XS, FMB,  FMS, FMOUT, 32, m0, 0, lane);
  } else {
    gemm_f32_out<1, 16, true >(XB, XS, FMW2, FMS, QOUT,  16, m0, 0, lane);
  }
  __syncthreads();
  //    deep L2: H2 = relu(H1 @ w2 + b2)   [64,128]x[128,64]; 2 of 4 N-tiles per wave
  gemm_relu_bf16<2, 4>(H1, H1S, W2T, W2S, b2, H2, H2S, m0, nh * 2, lane);
  __syncthreads();
  //    deep L3: H3 = relu(H2 @ w3 + b3)   [64,64]x[64,32]; 1 of 2 N-tiles per wave
  gemm_relu_bf16<1, 2>(H2, H2S, W3T, W3S, b3, H3, H3S, m0, nh, lane);
  __syncthreads();

  // 4) per-row epilogue: FM second order + linear + deep head + sigmoid
  if (tid < MT) {
    const int row = tid;
    const float* SFB  = wsf;
    const float* SFB2 = wsf + 16;
    float fmsec = 0.0f;
#pragma unroll
    for (int d = 0; d < 16; ++d) {
      float S  = FMOUT[row * 32 + d] + SFB[d];       // sum_emb[d]
      float P2 = FMOUT[row * 32 + 16 + d];           // x @ (2 fm_w fm_b)
      float Q  = QOUT[row * 16 + d];                 // x^2 @ fm_w^2
      fmsec += S * S - (Q + P2 + SFB2[d]);
    }
    fmsec *= 0.5f;
    float lin = LINP[4 * row] + LINP[4 * row + 1] + LINP[4 * row + 2] + LINP[4 * row + 3] + lin_b[0];
    float deep = bo[0];
#pragma unroll
    for (int j = 0; j < 32; ++j) deep = fmaf(bf2f(H3[row * H3S + j]), wo[j], deep);
    float z = lin + fmsec + deep + bias0[0];
    out[row0 + row] = 1.0f / (1.0f + __expf(-z));
  }
}

extern "C" void kernel_launch(void* const* d_in, const int* in_sizes, int n_in,
                              void* d_out, int out_size, void* d_ws, size_t ws_size,
                              hipStream_t stream) {
  (void)in_sizes; (void)n_in; (void)out_size; (void)ws_size;
  const float* x     = (const float*)d_in[0];
  const float* fm_w  = (const float*)d_in[1];
  const float* fm_b  = (const float*)d_in[2];
  const float* lin_w = (const float*)d_in[3];
  const float* lin_b = (const float*)d_in[4];
  const float* w1    = (const float*)d_in[5];
  const float* b1    = (const float*)d_in[6];
  const float* w2    = (const float*)d_in[7];
  const float* b2    = (const float*)d_in[8];
  const float* w3    = (const float*)d_in[9];
  const float* b3    = (const float*)d_in[10];
  const float* wo    = (const float*)d_in[11];
  const float* bo    = (const float*)d_in[12];
  const float* bias0 = (const float*)d_in[13];

  u16*   wsb = (u16*)d_ws;
  float* wsf = (float*)((char*)d_ws + (size_t)WS_BF_TOTAL * 2);
  float* out = (float*)d_out;

  deepfm_prep<<<(WS_BF_TOTAL + 255) / 256, 256, 0, stream>>>(fm_w, fm_b, w1, w2, w3, wsb, wsf);
  deepfm_main<<<B_ROWS / MT, TPB, SMEM_BYTES, stream>>>(
      x, lin_w, lin_b, b1, b2, b3, wo, bo, bias0, wsb, wsf, out);
}
